// DualIntentRecModel_83537113907244
// MI455X (gfx1250) — matrix operand order repaired and checked by reading, not compiled, not run on hardware
//
#include <hip/hip_runtime.h>
#include <hip/hip_bf16.h>
#include <math.h>

// ---------------- model constants ----------------
#define NUM_USER   100000
#define NUM_ITEM   50000
#define NUM_GROUP  20000
#define EMB        64
#define BATCH      1024
#define E_UI       2000000
#define E_U        1000000
#define E_I        1000000
#define INV_TEMP   10.0f
#define EPS_N      1e-12f

#define N_UG  (NUM_USER + NUM_GROUP)   // 120000
#define N_IG  (NUM_ITEM + NUM_GROUP)   // 70000
#define N_UI  (NUM_USER + NUM_ITEM)    // 150000

typedef _Float16 v16h __attribute__((ext_vector_type(16)));
typedef float    v8f  __attribute__((ext_vector_type(8)));

// ---------------- utility: zero ----------------
__global__ void zero_f32(float* __restrict__ p, size_t n) {
  size_t i = (size_t)blockIdx.x * blockDim.x + threadIdx.x;
  size_t stride = (size_t)gridDim.x * blockDim.x;
  for (; i < n; i += stride) p[i] = 0.0f;
}

// ---------------- CSR build (col-indexed) ----------------
__global__ void count_cols(const int* __restrict__ col, int* __restrict__ counts,
                           int nE) {
  int e = blockIdx.x * blockDim.x + threadIdx.x;
  if (e < nE) atomicAdd(&counts[col[e]], 1);
}

// Single-block exclusive scan; counts[] is rewritten in place with the
// exclusive prefix (becomes the fill cursor), offsets[] gets n+1 entries.
__global__ void exscan_kernel(int* __restrict__ counts, int* __restrict__ offsets,
                              int n) {
  __shared__ int sdata[1024];
  __shared__ int s_carry;
  int tid = threadIdx.x;
  if (tid == 0) s_carry = 0;
  __syncthreads();
  for (int base = 0; base < n; base += 1024) {
    int i = base + tid;
    int x = (i < n) ? counts[i] : 0;
    sdata[tid] = x;
    __syncthreads();
    for (int off = 1; off < 1024; off <<= 1) {
      int y = (tid >= off) ? sdata[tid - off] : 0;
      __syncthreads();
      sdata[tid] += y;
      __syncthreads();
    }
    int inc   = sdata[tid];      // inclusive within chunk
    int total = sdata[1023];
    int ex    = s_carry + inc - x;
    if (i < n) { offsets[i] = ex; counts[i] = ex; }
    __syncthreads();
    if (tid == 0) s_carry += total;
    __syncthreads();
  }
  if (tid == 0) offsets[n] = s_carry;
}

__global__ void csr_fill(const int* __restrict__ col, int* __restrict__ cursor,
                         int* __restrict__ eidx, int nE) {
  int e = blockIdx.x * blockDim.x + threadIdx.x;
  if (e < nE) {
    int slot = atomicAdd(&cursor[col[e]], 1);
    eidx[slot] = e;
  }
}

// ---------------- CSR propagation (gather, no atomics) ----------------
// out[c][:] = sum_{e in col-bucket c} in(row[e])[:] * val[e]
// One wave per output row; lane holds 2 floats (64 = 32*2).
__global__ void csr_prop(const float* __restrict__ p0, int n0,
                         const float* __restrict__ p1,
                         const int* __restrict__ row,
                         const float* __restrict__ val,
                         const int* __restrict__ offsets,
                         const int* __restrict__ eidx,
                         float* __restrict__ out, int nRows) {
  int wid = (blockIdx.x * blockDim.x + threadIdx.x) >> 5;
  if (wid >= nRows) return;
  int lane = threadIdx.x & 31;
  int beg = offsets[wid], end = offsets[wid + 1];
  float ax = 0.0f, ay = 0.0f;
  for (int k = beg; k < end; ++k) {
    int e = eidx[k];
    int r = row[e];
    float v = val[e];
    const float* src = (r < n0) ? (p0 + (size_t)r * EMB)
                                : (p1 + (size_t)(r - n0) * EMB);
    float2 d = *(const float2*)(src + lane * 2);
    ax += d.x * v;
    ay += d.y * v;
  }
  float2 o; o.x = ax; o.y = ay;
  *(float2*)(out + (size_t)wid * EMB + lane * 2) = o;
}

// ui_final = (ui_concat + agg) * 0.5, in-place on agg
__global__ void ui_final_kernel(const float* __restrict__ uei,
                                const float* __restrict__ item,
                                float* __restrict__ agg) {
  int gid = blockIdx.x * blockDim.x + threadIdx.x;
  if (gid >= N_UI * EMB) return;
  int r = gid >> 6, c = gid & 63;
  float uiv = (r < NUM_USER) ? uei[gid]
                             : item[(size_t)(r - NUM_USER) * EMB + c];
  agg[gid] = (uiv + agg[gid]) * 0.5f;
}

// ---------------- output gathers ----------------
__global__ void gather_out(const float* __restrict__ ug,   // 120000 x 64
                           const float* __restrict__ uif,  // 150000 x 64
                           const float* __restrict__ ig,   // 70000 x 64
                           const int* __restrict__ u,
                           const int* __restrict__ p,
                           const int* __restrict__ ng,
                           float* __restrict__ out) {
  int gid = blockIdx.x * blockDim.x + threadIdx.x;
  if (gid >= BATCH * EMB) return;
  int b = gid >> 6, c = gid & 63;
  const size_t PO = (size_t)BATCH * 128;
  const size_t NO = 2 * PO;
  out[(size_t)b * 128 + c]           = ug [(size_t)u[b] * EMB + c];
  out[(size_t)b * 128 + 64 + c]      = uif[(size_t)u[b] * EMB + c];
  out[PO + (size_t)b * 128 + c]      = ug [(size_t)(NUM_USER + p[b])  * EMB + c];
  out[PO + (size_t)b * 128 + 64 + c] = ig [(size_t)(NUM_ITEM + p[b])  * EMB + c];
  out[NO + (size_t)b * 128 + c]      = ug [(size_t)(NUM_USER + ng[b]) * EMB + c];
  out[NO + (size_t)b * 128 + 64 + c] = ig [(size_t)(NUM_ITEM + ng[b]) * EMB + c];
}

__global__ void reg_kernel(const float* __restrict__ ued,
                           const float* __restrict__ uei,
                           const float* __restrict__ ged,
                           const float* __restrict__ gei,
                           const int* __restrict__ u,
                           const int* __restrict__ p,
                           const int* __restrict__ ng,
                           float* __restrict__ reg_out) {
  int gid = blockIdx.x * blockDim.x + threadIdx.x;
  if (gid >= BATCH * EMB) return;
  int b = gid >> 6, c = gid & 63;
  size_t uo = (size_t)u[b]  * EMB + c;
  size_t po = (size_t)p[b]  * EMB + c;
  size_t no = (size_t)ng[b] * EMB + c;
  float s = uei[uo]*uei[uo] + ued[uo]*ued[uo]
          + gei[po]*gei[po] + ged[po]*ged[po]
          + gei[no]*gei[no] + ged[no]*ged[no];
  #pragma unroll
  for (int off = 1; off < 32; off <<= 1) s += __shfl_xor(s, off, 32);
  if ((threadIdx.x & 31) == 0) atomicAdd(reg_out, s * (0.5f / (float)BATCH));
}

// ---------------- normalize rows to f16 ----------
__global__ void normalize_rows_f16(const float* __restrict__ in,
                                   _Float16* __restrict__ out, int nrows) {
  int gid = blockIdx.x * blockDim.x + threadIdx.x;
  int r = gid >> 5;
  if (r >= nrows) return;
  int t = gid & 31;
  size_t base = (size_t)r * EMB + 2 * t;
  float x0 = in[base], x1 = in[base + 1];
  float ss = x0 * x0 + x1 * x1;
  #pragma unroll
  for (int off = 1; off < 32; off <<= 1) ss += __shfl_xor(ss, off, 32);
  float scale = 1.0f / fmaxf(sqrtf(ss), EPS_N);
  out[base]     = (_Float16)(x0 * scale);
  out[base + 1] = (_Float16)(x1 * scale);
}

// ---------------- SSL: tot_i = sum_j exp(na_i . nall_j / T) via WMMA -------
#define GEMM_WAVES 8
#define GEMM_TJ    16

__device__ __forceinline__ void load_btile(const _Float16* __restrict__ B16,
                                           int j0, int m, int kbB,
                                           v16h& b0, v16h& b1) {
  const _Float16* brow = B16 + (size_t)(j0 + m) * EMB + kbB;
  #pragma unroll
  for (int e = 0; e < 8; ++e) {
    b0[e] = brow[e]; b0[8 + e] = brow[16 + e];
    b1[e] = brow[32 + e]; b1[8 + e] = brow[48 + e];
  }
}

__global__ void ssl_gemm_expsum(const _Float16* __restrict__ A16,  // full A table
                                const _Float16* __restrict__ B16,  // full nall table
                                const int* __restrict__ idx,       // batch row ids
                                float* __restrict__ tot,           // [BATCH]
                                int Nall) {
  int lane = threadIdx.x & 31;
  int wave = threadIdx.x >> 5;
  int m0   = blockIdx.x * 16;
  int m    = lane & 15;
  int hi   = lane >> 4;

  // A: 16x32 f16 tile layout — lane half selects K sub-range (ISA 7.12.2)
  const _Float16* arow = A16 + (size_t)idx[m0 + m] * EMB;
  int kbA = hi ? 8 : 0;
  v16h a0, a1;
  #pragma unroll
  for (int e = 0; e < 8; ++e) {
    a0[e]     = arow[kbA + e];
    a0[8 + e] = arow[16 + kbA + e];
    a1[e]     = arow[32 + kbA + e];
    a1[8 + e] = arow[48 + kbA + e];
  }

  float s[8];
  #pragma unroll
  for (int v = 0; v < 8; ++v) s[v] = 0.0f;

  int jt0 = (blockIdx.y * GEMM_WAVES + wave) * GEMM_TJ;
  int kbB = hi ? 8 : 0;
  int j0  = jt0 * 16;

  if (j0 < Nall) {                               // wave-uniform condition
    v16h b0, b1;
    load_btile(B16, j0, m, kbB, b0, b1);
    for (int t = 0; t < GEMM_TJ; ++t) {
      int  j1   = j0 + 16;
      bool more = (t + 1 < GEMM_TJ) && (j1 < Nall);  // wave-uniform
      v16h nb0, nb1;
      if (more) load_btile(B16, j1, m, kbB, nb0, nb1);  // prefetch next tile
      v8f c = {};
      c = __builtin_amdgcn_wmma_f32_16x16x32_f16(false, a0, false, b0,
                                                 (short)0, c, false, false);
      c = __builtin_amdgcn_wmma_f32_16x16x32_f16(false, a1, false, b1,
                                                 (short)0, c, false, false);
      #pragma unroll
      for (int v = 0; v < 8; ++v) s[v] += __expf(c[v] * INV_TEMP);
      if (!more) break;
      b0 = nb0; b1 = nb1; j0 = j1;
    }
  }
  // reduce over N (16 lanes per half hold distinct cols of the same row)
  #pragma unroll
  for (int v = 0; v < 8; ++v) {
    #pragma unroll
    for (int off = 1; off < 16; off <<= 1) s[v] += __shfl_xor(s[v], off, 32);
  }
  if ((lane & 15) == 0) {
    #pragma unroll
    for (int v = 0; v < 8; ++v)
      atomicAdd(&tot[m0 + v + 8 * hi], s[v]);
  }
}

// loss += -(dot(na_i, nb_i)/T - log(tot_i)) / BATCH
__global__ void ssl_loss(const _Float16* __restrict__ A16,
                         const _Float16* __restrict__ B16,
                         const int* __restrict__ idx,
                         const float* __restrict__ tot,
                         float* __restrict__ ssl_out) {
  int gid = blockIdx.x * blockDim.x + threadIdx.x;
  int i = gid >> 5;
  if (i >= BATCH) return;
  int t = gid & 31;
  size_t ro = (size_t)idx[i] * EMB + 2 * t;
  float d = (float)A16[ro] * (float)B16[ro]
          + (float)A16[ro + 1] * (float)B16[ro + 1];
  #pragma unroll
  for (int off = 1; off < 32; off <<= 1) d += __shfl_xor(d, off, 32);
  if (t == 0) {
    float li = -(d * INV_TEMP - logf(tot[i])) * (1.0f / (float)BATCH);
    atomicAdd(ssl_out, li);
  }
}

// ---------------- launcher ----------------
extern "C" void kernel_launch(void* const* d_in, const int* in_sizes, int n_in,
                              void* d_out, int out_size, void* d_ws, size_t ws_size,
                              hipStream_t stream) {
  (void)in_sizes; (void)n_in; (void)out_size; (void)ws_size;
  const float* ued  = (const float*)d_in[0];   // user_emb_distinct
  const float* uei  = (const float*)d_in[1];   // user_emb_interest
  const float* ged  = (const float*)d_in[2];   // group_emb_distinct
  const float* gei  = (const float*)d_in[3];   // group_emb_interest
  const float* item = (const float*)d_in[4];   // item_emb
  const int*   hu_r = (const int*)d_in[5];
  const int*   hu_c = (const int*)d_in[6];
  const float* hu_v = (const float*)d_in[7];
  const int*   hi_r = (const int*)d_in[8];
  const int*   hi_c = (const int*)d_in[9];
  const float* hi_v = (const float*)d_in[10];
  const int*   ui_r = (const int*)d_in[11];
  const int*   ui_c = (const int*)d_in[12];
  const float* ui_v = (const float*)d_in[13];
  const int*   uin  = (const int*)d_in[14];
  const int*   pos  = (const int*)d_in[15];
  const int*   neg  = (const int*)d_in[16];
  float* out = (float*)d_out;

  // workspace layout
  float* W = (float*)d_ws;
  size_t off = 0;
  float* ug_b = W + off; off += (size_t)N_UG * EMB;
  float* ug_a = W + off; off += (size_t)N_UG * EMB;
  float* ig_b = W + off; off += (size_t)N_IG * EMB;
  float* ig_a = W + off; off += (size_t)N_IG * EMB;
  float* agg  = W + off; off += (size_t)N_UI * EMB;
  float* tot  = W + off; off += 4 * BATCH;
  int* counts  = (int*)(W + off); off += N_UI;        // max n = 150000
  int* offsets = (int*)(W + off); off += N_UI + 1;
  int* eidx    = (int*)(W + off); off += E_UI;        // max E = 2,000,000
  off = (off + 7) & ~(size_t)7;
  _Float16* H = (_Float16*)(W + off);
  _Float16* ug_user_n16  = H;                                 // 100000x64
  _Float16* ui_user_n16  = H + (size_t)NUM_USER * EMB;        // 100000x64
  _Float16* ug_group_n16 = H + (size_t)NUM_USER * EMB * 2;    // 20000x64
  _Float16* ig_group_n16 = ug_group_n16 + (size_t)NUM_GROUP * EMB;

  const int TB = 256;
  #define CDIV(a, b) (((a) + (b) - 1) / (b))

  // 0) zero small accumulators only (CSR gather writes every row, no big memset)
  zero_f32<<<16, TB, 0, stream>>>(tot, 4 * BATCH);
  zero_f32<<<1, 32, 0, stream>>>(out + (size_t)3 * BATCH * 128, 2);

  // ---- user-group hypergraph: build CSR once, 2 gather layers ----
  zero_f32<<<CDIV(N_UG, TB), TB, 0, stream>>>((float*)counts, N_UG);
  count_cols<<<CDIV(E_U, TB), TB, 0, stream>>>(hu_c, counts, E_U);
  exscan_kernel<<<1, 1024, 0, stream>>>(counts, offsets, N_UG);
  csr_fill<<<CDIV(E_U, TB), TB, 0, stream>>>(hu_c, counts, eidx, E_U);
  csr_prop<<<CDIV(N_UG * 32, TB), TB, 0, stream>>>(
      ued, NUM_USER, ged, hu_r, hu_v, offsets, eidx, ug_b, N_UG);
  csr_prop<<<CDIV(N_UG * 32, TB), TB, 0, stream>>>(
      ug_b, 1 << 30, nullptr, hu_r, hu_v, offsets, eidx, ug_a, N_UG);

  // ---- item-group hypergraph: build CSR once, 2 gather layers ----
  zero_f32<<<CDIV(N_IG, TB), TB, 0, stream>>>((float*)counts, N_IG);
  count_cols<<<CDIV(E_I, TB), TB, 0, stream>>>(hi_c, counts, E_I);
  exscan_kernel<<<1, 1024, 0, stream>>>(counts, offsets, N_IG);
  csr_fill<<<CDIV(E_I, TB), TB, 0, stream>>>(hi_c, counts, eidx, E_I);
  csr_prop<<<CDIV(N_IG * 32, TB), TB, 0, stream>>>(
      item, NUM_ITEM, gei, hi_r, hi_v, offsets, eidx, ig_b, N_IG);
  csr_prop<<<CDIV(N_IG * 32, TB), TB, 0, stream>>>(
      ig_b, 1 << 30, nullptr, hi_r, hi_v, offsets, eidx, ig_a, N_IG);

  // ---- user-item graph: 1 gather layer + (ui + agg)*0.5 ----
  zero_f32<<<CDIV(N_UI, TB), TB, 0, stream>>>((float*)counts, N_UI);
  count_cols<<<CDIV(E_UI, TB), TB, 0, stream>>>(ui_c, counts, E_UI);
  exscan_kernel<<<1, 1024, 0, stream>>>(counts, offsets, N_UI);
  csr_fill<<<CDIV(E_UI, TB), TB, 0, stream>>>(ui_c, counts, eidx, E_UI);
  csr_prop<<<CDIV(N_UI * 32, TB), TB, 0, stream>>>(
      uei, NUM_USER, item, ui_r, ui_v, offsets, eidx, agg, N_UI);
  ui_final_kernel<<<CDIV(N_UI * EMB, TB), TB, 0, stream>>>(uei, item, agg);

  // ---- output embeddings + reg ----
  gather_out<<<CDIV(BATCH * EMB, TB), TB, 0, stream>>>(
      ug_a, agg, ig_a, uin, pos, neg, out);
  reg_kernel<<<CDIV(BATCH * EMB, TB), TB, 0, stream>>>(
      ued, uei, ged, gei, uin, pos, neg, out + (size_t)3 * BATCH * 128);

  // ---- l2-normalize side tables to f16 ----
  normalize_rows_f16<<<CDIV(NUM_USER * 32, TB), TB, 0, stream>>>(
      ug_a, ug_user_n16, NUM_USER);                              // user_g_side
  normalize_rows_f16<<<CDIV(NUM_USER * 32, TB), TB, 0, stream>>>(
      agg, ui_user_n16, NUM_USER);                               // user_i_side
  normalize_rows_f16<<<CDIV(NUM_GROUP * 32, TB), TB, 0, stream>>>(
      ug_a + (size_t)NUM_USER * EMB, ug_group_n16, NUM_GROUP);   // group_u_side
  normalize_rows_f16<<<CDIV(NUM_GROUP * 32, TB), TB, 0, stream>>>(
      ig_a + (size_t)NUM_ITEM * EMB, ig_group_n16, NUM_GROUP);   // group_i_side

  // ---- four SSL terms: WMMA logits + exp row-sums, then loss reductions ----
  const int perBlk = GEMM_WAVES * GEMM_TJ;
  int ntU = NUM_USER / 16, ntG = NUM_GROUP / 16;
  dim3 gU(BATCH / 16, CDIV(ntU, perBlk));
  dim3 gG(BATCH / 16, CDIV(ntG, perBlk));
  ssl_gemm_expsum<<<gU, TB, 0, stream>>>(ui_user_n16, ug_user_n16, uin, tot + 0,         NUM_USER);
  ssl_gemm_expsum<<<gU, TB, 0, stream>>>(ug_user_n16, ui_user_n16, uin, tot + BATCH,     NUM_USER);
  ssl_gemm_expsum<<<gG, TB, 0, stream>>>(ig_group_n16, ug_group_n16, pos, tot + 2*BATCH, NUM_GROUP);
  ssl_gemm_expsum<<<gG, TB, 0, stream>>>(ug_group_n16, ig_group_n16, pos, tot + 3*BATCH, NUM_GROUP);

  float* ssl_slot = out + (size_t)3 * BATCH * 128 + 1;
  int lossBlocks = CDIV(BATCH * 32, TB);
  ssl_loss<<<lossBlocks, TB, 0, stream>>>(ui_user_n16, ug_user_n16, uin, tot + 0,         ssl_slot);
  ssl_loss<<<lossBlocks, TB, 0, stream>>>(ug_user_n16, ui_user_n16, uin, tot + BATCH,     ssl_slot);
  ssl_loss<<<lossBlocks, TB, 0, stream>>>(ig_group_n16, ug_group_n16, pos, tot + 2*BATCH, ssl_slot);
  ssl_loss<<<lossBlocks, TB, 0, stream>>>(ug_group_n16, ig_group_n16, pos, tot + 3*BATCH, ssl_slot);
  #undef CDIV
}
